// GEN_autoEncoderGCN_Encoder_graph_zone0_56530359550896
// MI455X (gfx1250) — matrix operand order repaired and checked by reading, not compiled, not run on hardware
//
#include <hip/hip_runtime.h>
#include <hip/hip_bf16.h>
#include <math.h>
#include <stdint.h>

typedef __attribute__((ext_vector_type(16))) _Float16 v16h;
typedef __attribute__((ext_vector_type(8)))  _Float16 v8h;
typedef __attribute__((ext_vector_type(2)))  _Float16 v2h;
typedef __attribute__((ext_vector_type(8)))  float    v8f;

#define N_NODES 248
#define KPAD1   256   // padded node dim (K and N of GEMM1)
#define N2      180
#define N2P     192   // padded N of GEMM2 / K of GEMM3
#define N3      80
#define BN_EPS  1e-5f

#define WAVES    8             // waves per block
#define TPB      (WAVES * 32)  // 256 threads
#define WAVE_BUF 8192          // halves per wave: buf0(4096) + buf1(4096)
#define SLAB_H   8192          // halves per B slab buffer (16 KB)
#define LDS_BYTES ((WAVES * WAVE_BUF + 2 * SLAB_H) * 2)   // 160 KB

// ---------------- hardware tanh (TRANS unit) ----------------
__device__ __forceinline__ float hw_tanh(float v) {
#if __has_builtin(__builtin_amdgcn_tanhf)
    return __builtin_amdgcn_tanhf(v);
#elif __has_builtin(__builtin_amdgcn_tanh_f32)
    return __builtin_amdgcn_tanh_f32(v);
#else
    float r;
    // ISA 7.4: 1 independent op / V_NOP after a TRANS op before use.
    asm volatile("v_tanh_f32 %0, %1\n\tv_nop" : "=v"(r) : "v"(v));
    return r;
#endif
}

// -------- CDNA5 async memory->LDS DMA (ASYNCcnt-tracked, no VGPR data) -----
__device__ __forceinline__ void async_copy16(const _Float16* gsrc,
                                             _Float16* ldst) {
    unsigned           lds = (unsigned)(uintptr_t)ldst;   // LDS addr = low 32b
    unsigned long long ga  = (unsigned long long)(uintptr_t)gsrc;
    asm volatile("global_load_async_to_lds_b128 %0, %1, off"
                 :: "v"(lds), "v"(ga) : "memory");
}

__device__ __forceinline__ void wait_async0() {
#if __has_builtin(__builtin_amdgcn_s_wait_asynccnt)
    __builtin_amdgcn_s_wait_asynccnt(0);
#else
    asm volatile("s_wait_asynccnt 0x0" ::: "memory");
#endif
}

// ---------------- prep kernels (one-off, tiny cost) ----------------

__global__ void k_zero(float* g, int n) {
    int i = blockIdx.x * blockDim.x + threadIdx.x;
    int s = gridDim.x * blockDim.x;
    for (; i < n; i += s) g[i] = 0.0f;
}

// Densify normalized adjacency: Gf32[n][s] = sum_{e: dst=n,src=s} dinv[s]*dinv[n]
// (+ dinv[n]^2 self loop). One thread per destination row -> deterministic.
__global__ void k_prep(const long long* __restrict__ ei, int nE,
                       float* __restrict__ Gf32) {
    __shared__ float dinv[256];
    int t = threadIdx.x;
    float di = 0.0f;
    if (t < N_NODES) {
        int cnt = 1; // self loop
        for (int e = 0; e < nE; ++e) cnt += ((int)ei[nE + e] == t);
        di = rsqrtf((float)cnt);
    }
    dinv[t] = di;
    __syncthreads();
    if (t < N_NODES) {
        float dn = dinv[t];
        float* row = Gf32 + t * KPAD1;   // row n (dst), cols = src s
        for (int e = 0; e < nE; ++e) {
            if ((int)ei[nE + e] == t) {
                int s = (int)ei[e];
                row[s] += dinv[s] * dn;
            }
        }
        row[t] += dn * dn;
    }
}

// Pack B matrices to f16, column-major-by-N (row n holds all K contiguously),
// zero-padded so padded K columns contribute nothing.
__global__ void k_pack(const float* __restrict__ Gf32,
                       const float* __restrict__ Wg,
                       const float* __restrict__ W2,
                       const float* __restrict__ W3,
                       _Float16* __restrict__ Gcm,
                       _Float16* __restrict__ W2p,
                       _Float16* __restrict__ W3p) {
    int i0 = blockIdx.x * blockDim.x + threadIdx.x;
    int st = gridDim.x * blockDim.x;
    float w = Wg[0];
    for (int i = i0; i < KPAD1 * KPAD1; i += st)
        Gcm[i] = (_Float16)(w * Gf32[i]);
    for (int i = i0; i < N2P * KPAD1; i += st) {
        int o = i >> 8, l = i & 255;
        W2p[i] = (o < N2 && l < N_NODES) ? (_Float16)W2[o * N_NODES + l]
                                         : (_Float16)0.0f;
    }
    for (int i = i0; i < N3 * N2P; i += st) {
        int o = i / N2P, l = i - o * N2P;
        W3p[i] = (l < N2) ? (_Float16)W3[o * N2 + l] : (_Float16)0.0f;
    }
}

// ---------------- fragment loaders ----------------

// A fragment (16x32 f16) from LDS row-major buffer, row stride S halves.
// lane<16: v[0..7]=K{k0..k0+7}, v[8..15]=K{k0+16..k0+23}; lane>=16: K base +8.
__device__ __forceinline__ v16h load_a(const _Float16* buf, int lane,
                                       int kTile, int S) {
    int m    = lane & 15;
    int koff = kTile * 32 + ((lane >> 4) << 3);
    const _Float16* p = buf + m * S + koff;
    v8h lo = *(const v8h*)(p);
    v8h hi = *(const v8h*)(p + 16);
    return __builtin_shufflevector(lo, hi, 0, 1, 2, 3, 4, 5, 6, 7,
                                   8, 9, 10, 11, 12, 13, 14, 15);
}

// B fragment (32x16 f16) from an LDS slab holding one 32-wide K slice,
// layout: column n -> 32 contiguous halves at slab[n*32].
// lane<16: col n=lane, K{0..7,16..23}; lane>=16: col n=lane-16, K{8..15,24..31}.
__device__ __forceinline__ v16h load_b_slab(const _Float16* slab, int lane,
                                            int nTile) {
    int n = nTile * 16 + (lane & 15);
    const _Float16* p = slab + n * 32 + ((lane >> 4) << 3);
    v8h lo = *(const v8h*)(p);
    v8h hi = *(const v8h*)(p + 16);
    return __builtin_shufflevector(lo, hi, 0, 1, 2, 3, 4, 5, 6, 7,
                                   8, 9, 10, 11, 12, 13, 14, 15);
}

// ---------------- pipelined shared-B GEMM stage ----------------
// kT-outer, per-lane accumulator array acc[NT]. The 32-wide K slab of B
// (totalChunks = Npad*4 16-byte chunks) is DMA'd by all TPB threads into a
// double-buffered LDS slab via global_load_async_to_lds_b128: the DMA for
// slab k+1 is issued before the WMMA burst on slab k; s_wait_asynccnt 0 +
// one barrier per kT close the loop. B fragments are consumed in pairs so
// each s_wait_dscnt covers two WMMAs.
template <int NT, int NKT, int NCH>
__device__ __forceinline__ void gemm_stage(const _Float16* __restrict__ Bsrc,
                                           int KS, int totalChunks,
                                           const _Float16* abuf, int astride,
                                           _Float16* slab, int tid, int lane,
                                           v8f acc[NT]) {
    // prologue: DMA slab for kT = 0
#pragma unroll
    for (int j = 0; j < NCH; ++j) {
        int c = tid + j * TPB;
        if (c < totalChunks) {
            int n = c >> 2, part = c & 3;
            async_copy16(Bsrc + (long long)n * KS + part * 8, slab + c * 8);
        }
    }
    wait_async0();
    __syncthreads();

#pragma unroll
    for (int kT = 0; kT < NKT; ++kT) {
        const int p = kT & 1;
        if (kT + 1 < NKT) {  // issue DMA for next slab (completion deferred)
            _Float16* nxt = slab + (1 - p) * SLAB_H;
#pragma unroll
            for (int j = 0; j < NCH; ++j) {
                int c = tid + j * TPB;
                if (c < totalChunks) {
                    int n = c >> 2, part = c & 3;
                    async_copy16(Bsrc + (long long)n * KS + (kT + 1) * 32 +
                                     part * 8,
                                 nxt + c * 8);
                }
            }
        }
        v16h a = load_a(abuf, lane, kT, astride);
        const _Float16* cur = slab + p * SLAB_H;
#pragma unroll
        for (int nT = 0; nT < NT; nT += 2) {
            v16h b0 = load_b_slab(cur, lane, nT);
            v16h b1 = b0;
            if (nT + 1 < NT) b1 = load_b_slab(cur, lane, nT + 1);
            acc[nT] = __builtin_amdgcn_wmma_f32_16x16x32_f16(
                false, a, false, b0, (short)0, acc[nT], false, false);
            if (nT + 1 < NT)
                acc[nT + 1] = __builtin_amdgcn_wmma_f32_16x16x32_f16(
                    false, a, false, b1, (short)0, acc[nT + 1], false, false);
        }
        if (kT + 1 < NKT) wait_async0();
        __syncthreads();
    }
}

// ---------------- fused main kernel ----------------
// 256 threads = 8 waves; each wave owns 16 batch rows (block: 128 rows).
// Dynamic LDS: 8 x (buf0 8KB | buf1 8KB) + 2 x 16KB shared B slabs = 160 KB
// -> 2 blocks per WGP (320 KB LDS), 4 waves per SIMD.

__global__ __launch_bounds__(TPB) void k_fused(
    const float* __restrict__ x,
    const _Float16* __restrict__ G,
    const _Float16* __restrict__ W2p,
    const _Float16* __restrict__ W3p,
    const float* __restrict__ bgcn, const float* __restrict__ gamma,
    const float* __restrict__ beta, const float* __restrict__ mean,
    const float* __restrict__ var,  const float* __restrict__ b2,
    const float* __restrict__ b3,   float* __restrict__ out) {
    extern __shared__ __align__(16) _Float16 lds[];

    int tid  = threadIdx.x;
    int lane = tid & 31;
    int wave = tid >> 5;
    long long row0 = ((long long)blockIdx.x * WAVES + wave) * 16;

    _Float16* buf0 = lds + wave * WAVE_BUF;
    _Float16* buf1 = buf0 + 4096;
    _Float16* slab = lds + WAVES * WAVE_BUF;

    // ---- stage 1: x[16x248] f32 -> f16 LDS [16][256], zero-padded ----
    for (int i = lane; i < 16 * 128; i += 32) {
        int r = i >> 7, p = i & 127;   // p = half2 column pair
        v2h h;
        if (p < 124) {
            const float2* xr = (const float2*)(x + (row0 + r) * N_NODES);
            float2 v = xr[p];
            h[0] = (_Float16)v.x;
            h[1] = (_Float16)v.y;
        } else {
            h[0] = (_Float16)0.0f;
            h[1] = (_Float16)0.0f;
        }
        *(v2h*)(buf0 + r * KPAD1 + p * 2) = h;
    }
    __syncthreads();

    float bg    = bgcn[0];
    float scale = gamma[0] * rsqrtf(var[0] + BN_EPS);
    float shift = beta[0] - mean[0] * scale;

    int nCol  = lane & 15;
    int mBase = (lane >> 4) << 3;

    // ---- GEMM1: h1 = tanh(x @ G + bg)*scale + shift  (N=256, K=256) ----
    {
        v8f z = {};
        v8f acc[16];
#pragma unroll
        for (int i = 0; i < 16; ++i) acc[i] = z;
        gemm_stage<16, 8, 4>(G, KPAD1, KPAD1 * 4, buf0, KPAD1, slab, tid, lane,
                             acc);
#pragma unroll
        for (int nT = 0; nT < 16; ++nT) {
            int n = nT * 16 + nCol;
#pragma unroll
            for (int r = 0; r < 8; ++r) {
                float t = hw_tanh(acc[nT][r] + bg) * scale + shift;
                float v = (n < N_NODES) ? t : 0.0f;   // zero the K-padding
                buf1[(mBase + r) * KPAD1 + n] = (_Float16)v;
            }
        }
    }
    __syncthreads();

    // ---- GEMM2: h2 = tanh(h1 @ W2^T + b2)  (N=192, K=256) ----
    {
        v8f z = {};
        v8f acc[12];
#pragma unroll
        for (int i = 0; i < 12; ++i) acc[i] = z;
        gemm_stage<12, 8, 3>(W2p, KPAD1, N2P * 4, buf1, KPAD1, slab, tid, lane,
                             acc);
#pragma unroll
        for (int nT = 0; nT < 12; ++nT) {
            int n = nT * 16 + nCol;
            float bias = (n < N2) ? b2[n] : 0.0f;
#pragma unroll
            for (int r = 0; r < 8; ++r) {
                float t = hw_tanh(acc[nT][r] + bias);
                float v = (n < N2) ? t : 0.0f;        // zero the K-padding
                buf0[(mBase + r) * N2P + n] = (_Float16)v;  // buf0 reused as h2
            }
        }
    }
    __syncthreads();

    // ---- GEMM3: out = h2 @ W3^T + b3  (N=80, K=192) ----
    {
        v8f z = {};
        v8f acc[5];
#pragma unroll
        for (int i = 0; i < 5; ++i) acc[i] = z;
        gemm_stage<5, 6, 2>(W3p, N2P, N3 * 4, buf0, N2P, slab, tid, lane, acc);
#pragma unroll
        for (int nT = 0; nT < 5; ++nT) {
            int n = nT * 16 + nCol;
            float bias = b3[n];
#pragma unroll
            for (int r = 0; r < 8; ++r)
                out[(row0 + mBase + r) * N3 + n] = acc[nT][r] + bias;
        }
    }
}

// ---------------- host launcher ----------------

extern "C" void kernel_launch(void* const* d_in, const int* in_sizes, int n_in,
                              void* d_out, int out_size, void* d_ws, size_t ws_size,
                              hipStream_t stream) {
    const float*     x   = (const float*)d_in[0];
    const long long* ei  = (const long long*)d_in[1];
    const float*     Wg  = (const float*)d_in[2];
    const float*     bg  = (const float*)d_in[3];
    const float*     gam = (const float*)d_in[4];
    const float*     bet = (const float*)d_in[5];
    const float*     mea = (const float*)d_in[6];
    const float*     var = (const float*)d_in[7];
    const float*     W2  = (const float*)d_in[8];
    const float*     b2  = (const float*)d_in[9];
    const float*     W3  = (const float*)d_in[10];
    const float*     b3  = (const float*)d_in[11];
    float* out = (float*)d_out;

    char* ws = (char*)d_ws;
    float*    Gf32 = (float*)ws;                        // 256 KB
    _Float16* Gcm  = (_Float16*)(ws + 262144);          // 128 KB
    _Float16* W2p  = (_Float16*)(ws + 262144 + 131072); // 96 KB
    _Float16* W3p  = (_Float16*)(ws + 262144 + 131072 + 98304); // 30 KB

    int nE = in_sizes[1] / 2;        // 2000 edges
    int B  = in_sizes[0] / N_NODES;  // 262144

    k_zero<<<256, 256, 0, stream>>>(Gf32, KPAD1 * KPAD1);
    k_prep<<<1, 256, 0, stream>>>(ei, nE, Gf32);
    k_pack<<<512, 256, 0, stream>>>(Gf32, Wg, W2, W3, Gcm, W2p, W3p);
    k_fused<<<B / (16 * WAVES), TPB, LDS_BYTES, stream>>>(
        x, Gcm, W2p, W3p, bg, gam, bet, mea, var, b2, b3, out);
}
